// DGIEncoder_35682588295724
// MI455X (gfx1250) — matrix operand order repaired and checked by reading, not compiled, not run on hardware
//
#include <hip/hip_runtime.h>

#define N_NODES 100000
#define N_EDGES 1600000
#define NUM_GRAPHS 1000
#define IN_DIM 256
#define HID_DIM 128
#define OUT_DIM 64

typedef __attribute__((ext_vector_type(16))) __bf16 v16bf;
typedef __attribute__((ext_vector_type(8)))  float  v8f;

union FragBF {
  v16bf v;
  uint4 q[2];
  unsigned short u[16];
};

__device__ __forceinline__ unsigned short f2bf(float f) {
  unsigned int u = __float_as_uint(f);
  return (unsigned short)((u + 0x7FFFu + ((u >> 16) & 1u)) >> 16);
}

// ---------------- utility kernels ----------------

__global__ void k_zero_f32(float* __restrict__ buf, int n) {
  int i = blockIdx.x * blockDim.x + threadIdx.x;
  if (i < n) buf[i] = 0.0f;
}

__global__ void k_deg_init(float* __restrict__ deg) {
  int i = blockIdx.x * blockDim.x + threadIdx.x;
  if (i < N_NODES) deg[i] = 1.0f;   // self-loop
}

__global__ void k_deg_count(const long long* __restrict__ ei, float* __restrict__ deg) {
  int e = blockIdx.x * blockDim.x + threadIdx.x;
  if (e < N_EDGES) atomicAdd(&deg[(int)ei[N_EDGES + e]], 1.0f);
}

__global__ void k_rsqrt(float* __restrict__ deg) {
  int i = blockIdx.x * blockDim.x + threadIdx.x;
  if (i < N_NODES) deg[i] = rsqrtf(deg[i]);
}

// fp32 -> bf16, 4 elements per thread
__global__ void k_cvt_x(const float* __restrict__ x, unsigned short* __restrict__ xb, int n4) {
  int i = blockIdx.x * blockDim.x + threadIdx.x;
  if (i >= n4) return;
  const float4 f = reinterpret_cast<const float4*>(x)[i];
  uint2 o;
  o.x = (unsigned)f2bf(f.x) | ((unsigned)f2bf(f.y) << 16);
  o.y = (unsigned)f2bf(f.z) | ((unsigned)f2bf(f.w) << 16);
  reinterpret_cast<uint2*>(xb)[i] = o;
}

// Shuffle W [K,N] row-major fp32 into per-lane-contiguous B-fragment bf16 order:
// wf[ ((t*KS + s)*32 + lane)*16 + e ] = W[ (s*32 + (lane>>4)*16 + e) * N + t*16 + (lane&15) ]
template <int NT, int KS>
__global__ void k_cvt_wfrag(const float* __restrict__ W, unsigned short* __restrict__ wf) {
  const int N = NT * 16;
  const int total = NT * 16 * KS * 32;  // == K*N
  int i = blockIdx.x * blockDim.x + threadIdx.x;
  if (i >= total) return;
  int e = i & 15;
  int l = (i >> 4) & 31;
  int ts = i >> 9;
  int s = ts % KS;
  int t = ts / KS;
  int krow = s * 32 + (l >> 4) * 16 + e;
  int col  = t * 16 + (l & 15);
  wf[i] = f2bf(W[(size_t)krow * N + col]);
}

// ---------------- WMMA GEMM: C[M,N] = A[M,K](bf16) * Bfrag (bf16) ----------------
// One wave per 16-row M tile, all NT column tiles, K in steps of 32.
template <int NT, int KS>
__global__ void k_gemm_bf16(const unsigned short* __restrict__ A,
                            const unsigned short* __restrict__ Bf,
                            float* __restrict__ C, int M) {
  const int K = KS * 32, N = NT * 16;
  int wave = (int)((blockIdx.x * blockDim.x + threadIdx.x) >> 5);
  int lane = threadIdx.x & 31;
  if (wave * 16 >= M) return;           // wave-uniform: EXEC stays all-ones
  int m  = wave * 16 + (lane & 15);
  int kb = (lane >> 4) * 16;
  const unsigned short* arow = A + (size_t)m * K + kb;

  v8f acc[NT] = {};
  for (int s = 0; s < KS; ++s) {
    FragBF a;
    const uint4* ap = reinterpret_cast<const uint4*>(arow + s * 32);
    a.q[0] = ap[0];       // K = kb + s*32 + 0..7
    a.q[1] = ap[1];       // K = kb + s*32 + 8..15
#pragma unroll
    for (int t = 0; t < NT; ++t) {
      FragBF b;
      const uint4* bp =
          reinterpret_cast<const uint4*>(Bf + ((size_t)(t * KS + s) * 32 + lane) * 16);
      b.q[0] = bp[0];
      b.q[1] = bp[1];
      acc[t] = __builtin_amdgcn_wmma_f32_16x16x32_bf16(
          false, a.v, false, b.v, (short)0, acc[t], false, false);
    }
  }
  // C/D layout: VGPR r -> row = wavetile + r + (lane/16)*8 ; col = t*16 + lane%16
  int r0 = wave * 16 + (lane >> 4) * 8;
  int c0 = lane & 15;
#pragma unroll
  for (int t = 0; t < NT; ++t) {
#pragma unroll
    for (int r = 0; r < 8; ++r) {
      C[(size_t)(r0 + r) * N + t * 16 + c0] = acc[t][r];
    }
  }
}

// ---------------- edge aggregation ----------------

// layer 1: 128 feats, one wave per edge, float4 per lane
__global__ void k_edge_agg1(const long long* __restrict__ ei, const float* __restrict__ dis,
                            const float* __restrict__ h1, float* __restrict__ agg) {
  long long gid = (long long)blockIdx.x * blockDim.x + threadIdx.x;
  int e = (int)(gid >> 5), lane = (int)(gid & 31);
  if (e >= N_EDGES) return;
  int s = (int)ei[e], d = (int)ei[N_EDGES + e];
  float coef = dis[s] * dis[d];
  const float4 v = reinterpret_cast<const float4*>(h1 + (size_t)s * HID_DIM)[lane];
  float* o = agg + (size_t)d * HID_DIM + lane * 4;
  atomicAdd(o + 0, v.x * coef);
  atomicAdd(o + 1, v.y * coef);
  atomicAdd(o + 2, v.z * coef);
  atomicAdd(o + 3, v.w * coef);
}

// layer 2: 64 feats, one wave per edge, float2 per lane
__global__ void k_edge_agg2(const long long* __restrict__ ei, const float* __restrict__ dis,
                            const float* __restrict__ h2, float* __restrict__ agg) {
  long long gid = (long long)blockIdx.x * blockDim.x + threadIdx.x;
  int e = (int)(gid >> 5), lane = (int)(gid & 31);
  if (e >= N_EDGES) return;
  int s = (int)ei[e], d = (int)ei[N_EDGES + e];
  float coef = dis[s] * dis[d];
  const float2 v = reinterpret_cast<const float2*>(h2 + (size_t)s * OUT_DIM)[lane];
  float* o = agg + (size_t)d * OUT_DIM + lane * 2;
  atomicAdd(o + 0, v.x * coef);
  atomicAdd(o + 1, v.y * coef);
}

// finalize layer 1: add self-loop + bias, ReLU, emit bf16 for GEMM2
__global__ void k_fin1(const float* __restrict__ agg, const float* __restrict__ h1,
                       const float* __restrict__ dis, const float* __restrict__ b1,
                       unsigned short* __restrict__ outb) {
  int i = blockIdx.x * blockDim.x + threadIdx.x;
  if (i >= N_NODES * HID_DIM) return;
  int node = i >> 7, c = i & 127;
  float di = dis[node];
  float v = agg[i] + h1[i] * di * di + b1[c];
  v = fmaxf(v, 0.0f);
  outb[i] = f2bf(v);
}

// finalize layer 2: add self-loop + bias (no relu), fp32 in place
__global__ void k_fin2(float* __restrict__ agg, const float* __restrict__ h2,
                       const float* __restrict__ dis, const float* __restrict__ b2) {
  int i = blockIdx.x * blockDim.x + threadIdx.x;
  if (i >= N_NODES * OUT_DIM) return;
  int node = i >> 6, c = i & 63;
  float di = dis[node];
  agg[i] = agg[i] + h2[i] * di * di + b2[c];
}

// ---------------- mean pool ----------------

__global__ void k_pool_add(const long long* __restrict__ batch, const float* __restrict__ out2,
                           float* __restrict__ ps, float* __restrict__ cnt) {
  long long gid = (long long)blockIdx.x * blockDim.x + threadIdx.x;
  int n = (int)(gid >> 5), lane = (int)(gid & 31);
  if (n >= N_NODES) return;
  int g = (int)batch[n];
  const float2 v = reinterpret_cast<const float2*>(out2 + (size_t)n * OUT_DIM)[lane];
  atomicAdd(&ps[g * OUT_DIM + lane * 2 + 0], v.x);
  atomicAdd(&ps[g * OUT_DIM + lane * 2 + 1], v.y);
  if (lane == 0) atomicAdd(&cnt[g], 1.0f);
}

__global__ void k_pool_div(const float* __restrict__ ps, const float* __restrict__ cnt,
                           float* __restrict__ out) {
  int i = blockIdx.x * blockDim.x + threadIdx.x;
  if (i >= NUM_GRAPHS * OUT_DIM) return;
  out[i] = ps[i] / fmaxf(cnt[i >> 6], 1.0f);
}

// ---------------- launch ----------------

extern "C" void kernel_launch(void* const* d_in, const int* in_sizes, int n_in,
                              void* d_out, int out_size, void* d_ws, size_t ws_size,
                              hipStream_t stream) {
  const float*     x     = (const float*)d_in[0];
  const float*     W1    = (const float*)d_in[1];
  const float*     b1    = (const float*)d_in[2];
  const float*     W2    = (const float*)d_in[3];
  const float*     b2    = (const float*)d_in[4];
  const long long* ei    = (const long long*)d_in[5];
  const long long* batch = (const long long*)d_in[6];
  float* out = (float*)d_out;

  char* ws = (char*)d_ws;
  // bufA: x_bf16 (51.2MB), later out1_bf16 (25.6MB)
  // bufB: h1 fp32 (51.2MB), later h2 fp32 (25.6MB)
  // bufC: agg1 fp32 (51.2MB), later agg2/out2 fp32 (25.6MB)
  constexpr size_t OFF_A   = 0;
  constexpr size_t OFF_B   = 51200000;
  constexpr size_t OFF_C   = 102400000;
  constexpr size_t OFF_DIS = 153600000;                 // 400,000 B
  constexpr size_t OFF_W1  = 154000384;                 // 65,536 B (frag bf16)
  constexpr size_t OFF_W2  = OFF_W1 + 65536;            // 16,384 B
  constexpr size_t OFF_PS  = OFF_W2 + 16384;            // 256,000 B
  constexpr size_t OFF_CNT = OFF_PS + 256000;           // 4,000 B

  unsigned short* xbf    = (unsigned short*)(ws + OFF_A);
  unsigned short* out1bf = (unsigned short*)(ws + OFF_A);
  float*          h      = (float*)(ws + OFF_B);        // h1 then h2
  float*          agg    = (float*)(ws + OFF_C);        // agg1 then agg2
  float*          dis    = (float*)(ws + OFF_DIS);      // deg -> rsqrt in place
  unsigned short* w1f    = (unsigned short*)(ws + OFF_W1);
  unsigned short* w2f    = (unsigned short*)(ws + OFF_W2);
  float*          ps     = (float*)(ws + OFF_PS);
  float*          cnt    = (float*)(ws + OFF_CNT);

  const int TB = 256;
  (void)in_sizes; (void)n_in; (void)out_size; (void)ws_size;

  // --- degrees / normalization ---
  k_deg_init<<<(N_NODES + TB - 1) / TB, TB, 0, stream>>>(dis);
  k_deg_count<<<(N_EDGES + TB - 1) / TB, TB, 0, stream>>>(ei, dis);
  k_rsqrt<<<(N_NODES + TB - 1) / TB, TB, 0, stream>>>(dis);

  // --- convert inputs to bf16 / fragment layouts ---
  k_cvt_x<<<(N_NODES * IN_DIM / 4 + TB - 1) / TB, TB, 0, stream>>>(x, xbf, N_NODES * IN_DIM / 4);
  k_cvt_wfrag<8, 8><<<(IN_DIM * HID_DIM + TB - 1) / TB, TB, 0, stream>>>(W1, w1f);
  k_cvt_wfrag<4, 4><<<(HID_DIM * OUT_DIM + TB - 1) / TB, TB, 0, stream>>>(W2, w2f);

  // --- layer 1: GEMM (WMMA bf16), scatter, finalize ---
  {
    int waves = N_NODES / 16;  // 6250, exact
    k_gemm_bf16<8, 8><<<(waves * 32 + TB - 1) / TB, TB, 0, stream>>>(xbf, w1f, h, N_NODES);
  }
  k_zero_f32<<<(N_NODES * HID_DIM + TB - 1) / TB, TB, 0, stream>>>(agg, N_NODES * HID_DIM);
  {
    long long thr = (long long)N_EDGES * 32;
    k_edge_agg1<<<(unsigned)((thr + TB - 1) / TB), TB, 0, stream>>>(ei, dis, h, agg);
  }
  k_fin1<<<(N_NODES * HID_DIM + TB - 1) / TB, TB, 0, stream>>>(agg, h, dis, b1, out1bf);

  // --- layer 2: GEMM (WMMA bf16), scatter, finalize ---
  {
    int waves = N_NODES / 16;
    k_gemm_bf16<4, 4><<<(waves * 32 + TB - 1) / TB, TB, 0, stream>>>(out1bf, w2f, h, N_NODES);
  }
  k_zero_f32<<<(N_NODES * OUT_DIM + TB - 1) / TB, TB, 0, stream>>>(agg, N_NODES * OUT_DIM);
  {
    long long thr = (long long)N_EDGES * 32;
    k_edge_agg2<<<(unsigned)((thr + TB - 1) / TB), TB, 0, stream>>>(ei, dis, h, agg);
  }
  k_fin2<<<(N_NODES * OUT_DIM + TB - 1) / TB, TB, 0, stream>>>(agg, h, dis, b2);

  // --- mean pool ---
  k_zero_f32<<<(NUM_GRAPHS * OUT_DIM + NUM_GRAPHS + TB - 1) / TB, TB, 0, stream>>>(
      ps, NUM_GRAPHS * OUT_DIM + NUM_GRAPHS);  // ps and cnt are contiguous
  {
    long long thr = (long long)N_NODES * 32;
    k_pool_add<<<(unsigned)((thr + TB - 1) / TB), TB, 0, stream>>>(batch, agg, ps, cnt);
  }
  k_pool_div<<<(NUM_GRAPHS * OUT_DIM + TB - 1) / TB, TB, 0, stream>>>(ps, cnt, out);
}
